// Model_26353919328446
// MI455X (gfx1250) — matrix operand order repaired and checked by reading, not compiled, not run on hardware
//
#include <hip/hip_runtime.h>

// ---------------------------------------------------------------------------
// tanh-RNN for MI455X (gfx1250), wave32 + WMMA f32_16x16x32_f16.
// Phase 1: x_proj = input @ W_ih^T + (b_ih+b_hh) -> written into d_out.
//          Full-N register blocking: input read exactly once (BW roofline).
// Phase 2: persistent 4-WG scan, h = tanh(xp + h @ W_hh^T), in-place in d_out,
//          cross-WG sync via agent-scope atomic grid barrier once per step.
// ---------------------------------------------------------------------------

typedef __attribute__((ext_vector_type(16))) _Float16 v16h;
typedef __attribute__((ext_vector_type(8)))  _Float16 v8h;
typedef __attribute__((ext_vector_type(8)))  float    v8f;

static constexpr int SEQ   = 2048;
static constexpr int BATCH = 64;
static constexpr int KDIM  = 256;   // input feature dim == hidden dim
static constexpr int HID   = 256;

static constexpr int SCAN_BLOCKS = 4;     // N=256 split into 4 chunks of 64
static constexpr int HB_PITCH    = 264;   // f16 pitch, h buffer  (528B rows, 16B aligned)
static constexpr int WT_PITCH    = 72;    // f16 pitch, W_hh^T chunk (144B rows)
static constexpr int WI_PITCH    = 264;   // f16 pitch, full W_ih^T (528B rows)

__device__ __forceinline__ v16h cat8(v8h lo, v8h hi) {
  return __builtin_shufflevector(lo, hi, 0,1,2,3,4,5,6,7,8,9,10,11,12,13,14,15);
}

// ---------------------------------------------------------------------------
// Generation-based grid barrier (participating blocks must be co-resident).
// state[0] = arrival counter, state[1] = generation.
// ---------------------------------------------------------------------------
__device__ __forceinline__ void grid_sync(unsigned* state, unsigned nblocks) {
  __threadfence();          // release: drain this thread's stores to L2
  __syncthreads();
  if (threadIdx.x == 0) {
    unsigned* cnt = state;
    unsigned* gen = state + 1;
    unsigned g = __hip_atomic_load(gen, __ATOMIC_RELAXED, __HIP_MEMORY_SCOPE_AGENT);
    unsigned a = __hip_atomic_fetch_add(cnt, 1u, __ATOMIC_ACQ_REL, __HIP_MEMORY_SCOPE_AGENT);
    if (a == nblocks - 1) {
      __hip_atomic_store(cnt, 0u, __ATOMIC_RELAXED, __HIP_MEMORY_SCOPE_AGENT);
      __hip_atomic_store(gen, g + 1u, __ATOMIC_RELEASE, __HIP_MEMORY_SCOPE_AGENT);
    } else {
      while (__hip_atomic_load(gen, __ATOMIC_ACQUIRE, __HIP_MEMORY_SCOPE_AGENT) == g) {
        __builtin_amdgcn_s_sleep(1);
      }
    }
  }
  __syncthreads();
  __threadfence();          // acquire: invalidate WGP$ so fresh h[s-1] is read
}

__global__ void init_barrier_kernel(unsigned* state) {
  state[0] = 0u;
  state[1] = 0u;
}

// ---------------------------------------------------------------------------
// Phase 1: out[m][n] = sum_k x[m][k] * W_ih[n][k] + b_ih[n] + b_hh[n]
// M = SEQ*BATCH = 131072, N = 256, K = 256.
// 512-thread WGs (16 waves). WG stages ALL of W_ih^T (256x256 f16, ~132KB LDS).
// Wave w owns M-tile (blockIdx*16 + w) and ALL 16 N-tiles: 16 f32 accumulators,
// one A fragment feeds 16 WMMAs per K-chunk -> input read exactly once.
// ---------------------------------------------------------------------------
__global__ __launch_bounds__(512) void xproj_kernel(
    const float* __restrict__ x,
    const float* __restrict__ Wih,
    const float* __restrict__ bih,
    const float* __restrict__ bhh,
    float* __restrict__ out)
{
  __shared__ _Float16 wt[KDIM * WI_PITCH];   // W_ih^T: [k][n], full 256 cols

  const int lane = threadIdx.x & 31;
  const int wave = threadIdx.x >> 5;         // 0..15
  const int half = lane >> 4;                // 0 or 1
  const int mloc = lane & 15;

  // Stage full W_ih^T into LDS (coalesced along k).
  for (int idx = threadIdx.x; idx < KDIM * KDIM; idx += 512) {
    const int n = idx >> 8;
    const int k = idx & 255;
    wt[k * WI_PITCH + n] = (_Float16)Wih[(size_t)n * KDIM + k];
  }
  __syncthreads();

  const int mtile = blockIdx.x * 16 + wave;
  const float* arow = x + (size_t)(mtile * 16 + mloc) * KDIM;

  // 16 accumulator tiles, initialized with the broadcast bias.
  v8f c[16];
#pragma unroll
  for (int nt = 0; nt < 16; ++nt) {
    const int ncol = nt * 16 + mloc;
    const float bsum = bih[ncol] + bhh[ncol];
#pragma unroll
    for (int r = 0; r < 8; ++r) c[nt][r] = bsum;
  }

  for (int kc = 0; kc < KDIM; kc += 32) {
    // A fragment: two contiguous 8-float runs from global, cvt to f16.
    v8h alo, ahi;
    const float* p0 = arow + kc + 8 * half;
    const float* p1 = arow + kc + 16 + 8 * half;
#pragma unroll
    for (int i = 0; i < 8; ++i) { alo[i] = (_Float16)p0[i]; ahi[i] = (_Float16)p1[i]; }
    const v16h a = cat8(alo, ahi);

    // One A fragment feeds all 16 N tiles.
    const _Float16* brow = &wt[(kc + 8 * half) * WI_PITCH + mloc];
#pragma unroll
    for (int nt = 0; nt < 16; ++nt) {
      v16h b;
#pragma unroll
      for (int i = 0; i < 8; ++i) {
        b[i]     = brow[i * WI_PITCH + nt * 16];
        b[8 + i] = brow[(16 + i) * WI_PITCH + nt * 16];
      }
      c[nt] = __builtin_amdgcn_wmma_f32_16x16x32_f16(false, a, false, b,
                                                     (short)0, c[nt], false, false);
    }
  }

  float* orow = out + (size_t)(mtile * 16) * HID;
#pragma unroll
  for (int r = 0; r < 8; ++r) {
    const size_t rbase = (size_t)(r + 8 * half) * HID;
#pragma unroll
    for (int nt = 0; nt < 16; ++nt) {
      orow[rbase + nt * 16 + mloc] = c[nt][r];
    }
  }
}

// ---------------------------------------------------------------------------
// Phase 2: persistent scan. 4 WGs x 256 threads (8 waves).
// WG b owns hidden columns [b*64, b*64+64). Wave w: mtile = w&3 (batch tile),
// N-tile pair (w>>2)*2 .. +1 — two accumulators sharing one A fragment gives
// two independent WMMA chains (no D->A/B hazard back-to-back).
// Per step: restage h_{s-1} (fp32 in d_out) into LDS as f16, GEMM, tanh,
// store fp32 in place, grid barrier.
// ---------------------------------------------------------------------------
__global__ __launch_bounds__(256) void scan_kernel(
    const float* __restrict__ Whh,
    float* __restrict__ out,
    unsigned* __restrict__ barrier_state)
{
  __shared__ _Float16 wt[KDIM * WT_PITCH];       // W_hh^T chunk: [k][nloc], 64 cols
  __shared__ _Float16 hbuf[BATCH * HB_PITCH];    // h_{s-1} as f16: [b][k]

  const int lane   = threadIdx.x & 31;
  const int wave   = threadIdx.x >> 5;           // 0..7
  const int mtile  = wave & 3;                   // batch tile (64/16)
  const int npair  = wave >> 2;                  // 0..1
  const int nbloc  = npair * 32;                 // local col base of tile pair
  const int nchunk = blockIdx.x * 64;            // this WG's hidden columns
  const int half   = lane >> 4;
  const int mloc   = lane & 15;

  // Stage W_hh^T chunk once (lives in LDS for all 2048 steps).
  for (int idx = threadIdx.x; idx < 64 * KDIM; idx += 256) {
    const int nloc = idx >> 8;
    const int k    = idx & 255;
    wt[k * WT_PITCH + nloc] = (_Float16)Whh[(size_t)(nchunk + nloc) * KDIM + k];
  }
  __syncthreads();

  const int ncol0 = nchunk + nbloc + mloc;       // lane's column, first tile
  const int ncol1 = ncol0 + 16;                  // lane's column, second tile
  const int mrow  = mtile * 16 + mloc;           // lane's A row (batch idx)

  for (int s = 0; s < SEQ; ++s) {
    float* outrow = out + (size_t)s * BATCH * HID;

    // Accumulators = x_proj tiles (still resident in d_out row s).
    v8f c0, c1;
#pragma unroll
    for (int r = 0; r < 8; ++r) {
      const size_t rbase = (size_t)(mtile * 16 + r + 8 * half) * HID;
      c0[r] = outrow[rbase + ncol0];
      c1[r] = outrow[rbase + ncol1];
    }

    if (s > 0) {
      const float* hprev = out + (size_t)(s - 1) * BATCH * HID;
      // Restage h_{s-1} into LDS as f16 (coalesced global reads).
      for (int idx = threadIdx.x; idx < BATCH * KDIM; idx += 256) {
        const int row = idx >> 8;
        const int col = idx & 255;
        hbuf[row * HB_PITCH + col] = (_Float16)hprev[idx];
      }
      __syncthreads();

#pragma unroll
      for (int kc = 0; kc < KDIM; kc += 32) {
        const _Float16* ar = &hbuf[mrow * HB_PITCH + kc + 8 * half];
        v8h alo = *(const v8h*)(ar);
        v8h ahi = *(const v8h*)(ar + 16);
        const v16h a = cat8(alo, ahi);

        const _Float16* br = &wt[(kc + 8 * half) * WT_PITCH + nbloc + mloc];
        v16h b0, b1;
#pragma unroll
        for (int i = 0; i < 8; ++i) {
          b0[i]     = br[i * WT_PITCH];
          b0[8 + i] = br[(16 + i) * WT_PITCH];
          b1[i]     = br[i * WT_PITCH + 16];
          b1[8 + i] = br[(16 + i) * WT_PITCH + 16];
        }

        c0 = __builtin_amdgcn_wmma_f32_16x16x32_f16(false, a, false, b0,
                                                    (short)0, c0, false, false);
        c1 = __builtin_amdgcn_wmma_f32_16x16x32_f16(false, a, false, b1,
                                                    (short)0, c1, false, false);
      }
    }

    // h_s = tanh(c); overwrite x_proj row s with hidden state (fp32 output).
#pragma unroll
    for (int r = 0; r < 8; ++r) {
      const size_t rbase = (size_t)(mtile * 16 + r + 8 * half) * HID;
      outrow[rbase + ncol0] = tanhf(c0[r]);
      outrow[rbase + ncol1] = tanhf(c1[r]);
    }

    grid_sync(barrier_state, SCAN_BLOCKS);   // also fences hbuf reuse
  }
}

// ---------------------------------------------------------------------------
extern "C" void kernel_launch(void* const* d_in, const int* in_sizes, int n_in,
                              void* d_out, int out_size, void* d_ws, size_t ws_size,
                              hipStream_t stream) {
  const float* x   = (const float*)d_in[0];   // [SEQ, BATCH, IN]
  const float* Wih = (const float*)d_in[1];   // [HID, IN]
  const float* Whh = (const float*)d_in[2];   // [HID, HID]
  const float* bih = (const float*)d_in[3];   // [HID]
  const float* bhh = (const float*)d_in[4];   // [HID]
  float* out = (float*)d_out;                 // [SEQ*BATCH, HID]
  unsigned* bar = (unsigned*)d_ws;

  (void)in_sizes; (void)n_in; (void)out_size; (void)ws_size;

  hipLaunchKernelGGL(init_barrier_kernel, dim3(1), dim3(1), 0, stream, bar);

  // 131072/16 = 8192 M tiles; 16 per WG -> 512 WGs, each covering full N.
  hipLaunchKernelGGL(xproj_kernel, dim3(512), dim3(512), 0, stream,
                     x, Wih, bih, bhh, out);

  hipLaunchKernelGGL(scan_kernel, dim3(SCAN_BLOCKS), dim3(256), 0, stream,
                     Whh, out, bar);
}
